// RegiNet_CrossViTv2_SW_84009560310409
// MI455X (gfx1250) — matrix operand order repaired and checked by reading, not compiled, not run on hardware
//
#include <hip/hip_runtime.h>
#include <hip/hip_bf16.h>

// ---------------------------------------------------------------------------
// Types for WMMA / TDM (CDNA5 / gfx1250, wave32)
// ---------------------------------------------------------------------------
typedef __attribute__((ext_vector_type(16))) _Float16 v16h;
typedef __attribute__((ext_vector_type(8)))  float    v8f;
typedef __attribute__((ext_vector_type(4)))  unsigned int u32x4;
typedef __attribute__((ext_vector_type(8)))  int      i32x8;
typedef __attribute__((ext_vector_type(4)))  int      i32x4;

#if defined(__has_builtin)
#if __has_builtin(__builtin_amdgcn_tensor_load_to_lds)
#define HAVE_TDM 1
#endif
#endif

#define GEMM_BIAS   1
#define GEMM_RES    2
#define GEMM_GELU   4
#define GEMM_TRANSB 8

#ifdef HAVE_TDM
// ---------------------------------------------------------------------------
// Tensor Data Mover: 2D tile load, f32 elements, tile = 32 (dim0) x 16 (dim1),
// packed row-major into LDS (row n -> LDS[n*32 + k]). TDM OOB reads return 0,
// which implements K/Cout zero-padding for free (ISA 08_async_tensor 8.2).
// D# packing per ISA 8.3/8.4 (group0 128b, group1 256b; groups 2/3 zero = 2D).
// ---------------------------------------------------------------------------
__device__ __forceinline__ void tdm_load_tile_2d(
    unsigned lds_byte_off, const float* gsrc,
    unsigned dim0, unsigned dim1, unsigned stride0)
{
  unsigned long long ga = (unsigned long long)(uintptr_t)gsrc;
  u32x4 g0;
  g0.x = 1u;                                   // count=1 (valid user D#)
  g0.y = lds_byte_off;                         // lds_addr
  g0.z = (unsigned)ga;                         // global_addr[31:0]
  g0.w = (unsigned)((ga >> 32) & 0x01ffffffu) | (2u << 30); // addr[56:32], type=2
  i32x8 g1;
  g1[0] = (int)(2u << 16);                     // data_size=2 -> 4-byte elements
  g1[1] = (int)((dim0 & 0xffffu) << 16);       // tensor_dim0[15:0] @ bits 63:48
  g1[2] = (int)(((dim0 >> 16) & 0xffffu) | ((dim1 & 0xffffu) << 16));
  g1[3] = (int)(((dim1 >> 16) & 0xffffu) | (32u << 16)); // tile_dim0 = 32
  g1[4] = 16;                                  // tile_dim1 = 16, tile_dim2 = 0
  g1[5] = (int)stride0;                        // tensor_dim0_stride[31:0]
  g1[6] = 0;
  g1[7] = 0;
  i32x4 z4 = {0, 0, 0, 0};
#if defined(__clang_major__) && (__clang_major__ >= 23)
  i32x8 z8 = {0, 0, 0, 0, 0, 0, 0, 0};
  __builtin_amdgcn_tensor_load_to_lds(g0, g1, z4, z4, z8, 0);
#else
  __builtin_amdgcn_tensor_load_to_lds(g0, g1, z4, z4, 0);
#endif
}
#endif

// ---------------------------------------------------------------------------
// Generic WMMA GEMM: C[M,N] = act(A[M,K] @ B[K,N] + bias) + Res
// fp32 in memory, f16 WMMA compute, f32 accumulate.
// Block = 128 threads = 4 waves; each wave owns a 16x16 C tile.
// B K-chunk staged TRANSPOSED in LDS (BsT[n][k]) so each lane's fragment is a
// contiguous 32B -> single v16h LDS read (ds_load_b128 x2, no strided gather).
// Full interior tiles take an unchecked (vectorizable) load path.
// ---------------------------------------------------------------------------
__global__ __launch_bounds__(128) void k_gemm(
    const float* __restrict__ A, int lda,
    const float* __restrict__ B, int ldb,
    const float* __restrict__ bias,
    const float* __restrict__ Res, int ldr,
    float* __restrict__ C, int ldc,
    int M, int N, int K, int flags)
{
  __shared__ _Float16 BsT[16 * 32];   // [n][k]
  const int wave = threadIdx.x >> 5;
  const int lane = threadIdx.x & 31;
  const int half = lane >> 4;         // 0: lanes 0-15, 1: lanes 16-31
  const int mrow = lane & 15;
  const int m0 = (blockIdx.x * 4 + wave) * 16;
  const int n0 = blockIdx.y * 16;

  v8f acc = {};
  const bool mFull = (m0 + 16) <= M;
  const bool nFull = (n0 + 16) <= N;
  const int  mA = (m0 + mrow) < M ? (m0 + mrow) : 0;
  const float* Arow = A + (size_t)mA * lda;

  const int sidx = threadIdx.x * 4;         // 128 thr * 4 elem = 512
  const int snn = sidx >> 5, sk4 = sidx & 31;

  for (int kc = 0; kc < K; kc += 32) {
    const bool kFull = (kc + 32) <= K;

    // ---- stage B chunk into BsT[n][k], zero padded on edges ----
    if (kFull && nFull) {
      if (flags & GEMM_TRANSB) {
        const float* src = B + (size_t)(n0 + snn) * ldb + kc + sk4;
#pragma unroll
        for (int e = 0; e < 4; ++e) BsT[snn * 32 + sk4 + e] = (_Float16)src[e];
      } else {
#pragma unroll
        for (int e = 0; e < 4; ++e)
          BsT[snn * 32 + sk4 + e] =
              (_Float16)B[(size_t)(kc + sk4 + e) * ldb + n0 + snn];
      }
    } else {
#pragma unroll
      for (int e = 0; e < 4; ++e) {
        int kg = kc + sk4 + e, ng = n0 + snn;
        float v = 0.f;
        if (kg < K && ng < N)
          v = (flags & GEMM_TRANSB) ? B[(size_t)ng * ldb + kg]
                                    : B[(size_t)kg * ldb + ng];
        BsT[snn * 32 + sk4 + e] = (_Float16)v;
      }
    }
    __syncthreads();

    if (kc + 32 < K) __builtin_prefetch(Arow + kc + 32, 0, 0); // global_prefetch_b8

    // ---- A fragment (contiguous K order k = 16*half + e, matching B):
    //      lane's K values are one contiguous 16-float run -> unchecked loads on interior tiles ----
    v16h a;
    if (mFull && kFull) {
      const float* Ap = Arow + kc + half * 16;
#pragma unroll
      for (int e = 0; e < 8; ++e) a[e] = (_Float16)Ap[e];
#pragma unroll
      for (int e = 0; e < 8; ++e) a[e + 8] = (_Float16)Ap[8 + e];
    } else {
      const int mOK = (m0 + mrow) < M;
#pragma unroll
      for (int e = 0; e < 16; ++e) {
        int kk = e + half * 16;
        int kg = kc + kk;
        float v = (mOK && kg < K) ? Arow[kg] : 0.f;
        a[e] = (_Float16)v;
      }
    }
    // ---- B fragment: contiguous 32B per lane ----
    v16h b = *(const v16h*)&BsT[mrow * 32 + half * 16];

    acc = __builtin_amdgcn_wmma_f32_16x16x32_f16(
        false, a, false, b, (short)0, acc, false, false);
    __syncthreads();
  }

  // ---- epilogue: C layout VGPR r -> M=m0+r(+8 upper half), N=n0+mrow ----
#pragma unroll
  for (int r = 0; r < 8; ++r) {
    int m = m0 + r + half * 8;
    int n = n0 + mrow;
    if (m < M && n < N) {
      float v = acc[r];
      if (flags & GEMM_BIAS) v += bias[n];
      if (flags & GEMM_GELU) v = 0.5f * v * (1.f + erff(v * 0.70710678f));
      if (flags & GEMM_RES)  v += Res[(size_t)m * ldr + n];
      C[(size_t)m * ldc + n] = v;
    }
  }
}

// ---------------------------------------------------------------------------
// Implicit-GEMM 3x3x3 conv3d ('SAME') on 128^3 volume, NCDHW.
// M = voxels (16 per wave), N = Cout (<=16, one tile), K = Cin*27.
// Weight panels are DMA'd by the Tensor Data Mover into a DOUBLE-BUFFERED LDS
// tile: chunk i+1's DMA is issued before chunk i is consumed, released with
// s_wait_tensorcnt<=1 (TDM ops complete in order per wave). The im2col gather
// and the WMMA both overlap the in-flight DMA.
// Fused bias + batchnorm + ReLU, or fused base-add (x3d = repeat(x,3)+h).
// ---------------------------------------------------------------------------
__global__ __launch_bounds__(32) void k_conv3d(
    const float* __restrict__ in, int Cin,
    const float* __restrict__ Wt,   // [Cout][Cin*27]
    const float* __restrict__ bias, int Cout,
    const float* __restrict__ bn_g, const float* __restrict__ bn_b,
    const float* __restrict__ bn_m, const float* __restrict__ bn_v,
    const float* __restrict__ base, // x volume (channel 0) or null
    float* __restrict__ out)
{
#ifdef HAVE_TDM
  __shared__ float Wsf[2][16 * 32];  // ping-pong [n][k] f32 tiles
#else
  __shared__ float Wsf[1][16 * 32];
#endif
  const int lane = threadIdx.x;
  const int half = lane >> 4;
  const int mrow = lane & 15;
  const int p0 = blockIdx.x * 16;
  const int K = Cin * 27;

  const int p = p0 + mrow;
  const int z = p >> 14, y = (p >> 7) & 127, x = p & 127;

#ifdef HAVE_TDM
  // prologue: DMA chunk 0 into buffer 0
  tdm_load_tile_2d((unsigned)(uintptr_t)&Wsf[0][0], Wt,
                   (unsigned)K, (unsigned)Cout, (unsigned)K);
#endif
  int buf = 0;

  v8f acc = {};
  for (int kc = 0; kc < K; kc += 32) {
#ifdef HAVE_TDM
    if (kc + 32 < K) {
      // LDS reads of the buffer being overwritten (read in the previous iteration) must drain
      asm volatile("s_wait_dscnt 0" ::: "memory");
      tdm_load_tile_2d((unsigned)(uintptr_t)&Wsf[buf ^ 1][0], Wt + kc + 32,
                       (unsigned)K, (unsigned)Cout, (unsigned)K);
    }
#else
    __syncthreads();
    for (int i = lane; i < 512; i += 32) {
      int kk = i & 31, n = i >> 5;
      int kg = kc + kk;
      Wsf[0][n * 32 + kk] = (kg < K && n < Cout) ? Wt[(size_t)n * K + kg] : 0.f;
    }
    __syncthreads();
#endif

    // ---- im2col A-fragment gather (overlaps the TDM DMA) ----
    v16h a;
#pragma unroll
    for (int e = 0; e < 16; ++e) {
      int kk = e + half * 16;
      int kg = kc + kk;
      float v = 0.f;
      if (kg < K) {
        int ci = kg / 27, r = kg % 27;
        int dz = r / 9 - 1, dy = (r % 9) / 3 - 1, dx = r % 3 - 1;
        int zz = z + dz, yy = y + dy, xx = x + dx;
        if (((unsigned)zz < 128u) && ((unsigned)yy < 128u) && ((unsigned)xx < 128u))
          v = in[((size_t)ci << 21) + ((size_t)zz << 14) + (yy << 7) + xx];
      }
      a[e] = (_Float16)v;
    }

#ifdef HAVE_TDM
    // release the older transfer only; next chunk's DMA stays in flight
    if (kc + 32 < K) __builtin_amdgcn_s_wait_tensorcnt(1);
    else             __builtin_amdgcn_s_wait_tensorcnt(0);
#endif
    // ---- B fragment: contiguous 64B of f32 per lane, convert to f16 ----
    v16h b;
    const float* wr = &Wsf[buf][mrow * 32 + half * 16];
#pragma unroll
    for (int e = 0; e < 16; ++e) b[e] = (_Float16)wr[e];

    acc = __builtin_amdgcn_wmma_f32_16x16x32_f16(
        false, a, false, b, (short)0, acc, false, false);
#ifdef HAVE_TDM
    buf ^= 1;
#endif
  }

#pragma unroll
  for (int r = 0; r < 8; ++r) {
    int pv = p0 + r + half * 8;  // voxel (M)
    int n  = mrow;               // channel (N)
    if (n < Cout) {
      float v = acc[r] + bias[n];
      if (bn_g) {
        float inv = bn_g[n] * rsqrtf(bn_v[n] + 1e-5f);
        v = (v - bn_m[n]) * inv + bn_b[n];
        v = v > 0.f ? v : 0.f;
      }
      if (base) v += base[pv];
      out[((size_t)n << 21) + pv] = v;
    }
  }
}

// ---------------------------------------------------------------------------
// Projector: Rodrigues pose + cone-beam ray march + trilinear sampling of x3d.
// ---------------------------------------------------------------------------
__global__ __launch_bounds__(128) void k_project(
    const float* __restrict__ vol,     // [3][128][128][128]
    const float* __restrict__ theta,   // [6]
    const float* __restrict__ corner,  // [8][4]
    const float* __restrict__ prm,     // src, det, pix, _
    float* __restrict__ x2d)           // [3][128][128]
{
  int idx = blockIdx.x * blockDim.x + threadIdx.x;
  if (idx >= 128 * 128) return;
  int hh = idx >> 7, ww = idx & 127;
  float src = prm[0], det = prm[1], pix = prm[2];

  float rx = theta[0], ry = theta[1], rz = theta[2];
  float tx = theta[3], ty = theta[4], tz = theta[5];
  float ang = sqrtf(rx * rx + ry * ry + rz * rz + 1e-12f);
  float ax = rx / ang, ay = ry / ang, az = rz / ang;
  float s = sinf(ang), co = cosf(ang), o = 1.f - co;

  float T[3][4];
  T[0][0] = 1.f - o * (ay * ay + az * az);
  T[0][1] = -s * az + o * ax * ay;
  T[0][2] =  s * ay + o * ax * az;
  T[1][0] =  s * az + o * ax * ay;
  T[1][1] = 1.f - o * (ax * ax + az * az);
  T[1][2] = -s * ax + o * ay * az;
  T[2][0] = -s * ay + o * ax * az;
  T[2][1] =  s * ax + o * ay * az;
  T[2][2] = 1.f - o * (ax * ax + ay * ay);
  T[0][3] = tx; T[1][3] = ty; T[2][3] = tz;

  float dmin = 1e30f, dmax = -1e30f;
#pragma unroll
  for (int n = 0; n < 8; ++n) {
    const float* cp = corner + n * 4;
    float c0 = 0.f, c1 = 0.f, c2 = 0.f;
#pragma unroll
    for (int k = 0; k < 4; ++k) {
      c0 += cp[k] * T[0][k];
      c1 += cp[k] * T[1][k];
      c2 += cp[k] * T[2][k];
    }
    float d = sqrtf(c0 * c0 + c1 * c1 + (c2 - src) * (c2 - src));
    dmin = fminf(dmin, d); dmax = fmaxf(dmax, d);
  }

  float dxv = (ww - 63.5f) * pix;
  float dyv = (hh - 63.5f) * pix;
  float dzv = -det;
  float invn = rsqrtf(dxv * dxv + dyv * dyv + dzv * dzv);
  dxv *= invn; dyv *= invn; dzv *= invn;

  float a0 = 0.f, a1 = 0.f, a2 = 0.f;
  for (int m = 0; m < 64; ++m) {
    float t = dmin + (dmax - dmin) * (m * (1.f / 63.f));
    float px = dxv * t, py = dyv * t, pz = dzv * t + src;
    float g0 = T[0][0] * px + T[0][1] * py + T[0][2] * pz + T[0][3];
    float g1 = T[1][0] * px + T[1][1] * py + T[1][2] * pz + T[1][3];
    float g2 = T[2][0] * px + T[2][1] * py + T[2][2] * pz + T[2][3];
    float fx = (g0 + 1.f) * 0.5f * 127.f;
    float fy = (g1 + 1.f) * 0.5f * 127.f;
    float fz = (g2 + 1.f) * 0.5f * 127.f;
    float x0 = floorf(fx), y0 = floorf(fy), z0 = floorf(fz);
    float wx = fx - x0, wy = fy - y0, wz = fz - z0;
#pragma unroll
    for (int oz = 0; oz < 2; ++oz)
#pragma unroll
      for (int oy = 0; oy < 2; ++oy)
#pragma unroll
        for (int ox = 0; ox < 2; ++ox) {
          float w = (oz ? wz : 1.f - wz) * (oy ? wy : 1.f - wy) * (ox ? wx : 1.f - wx);
          int xi = (int)fminf(fmaxf(x0 + ox, 0.f), 127.f);
          int yi = (int)fminf(fmaxf(y0 + oy, 0.f), 127.f);
          int zi = (int)fminf(fmaxf(z0 + oz, 0.f), 127.f);
          size_t off = ((size_t)zi << 14) + (yi << 7) + xi;
          a0 += vol[off] * w;
          a1 += vol[(1ull << 21) + off] * w;
          a2 += vol[(2ull << 21) + off] * w;
        }
  }
  x2d[idx] = a0;
  x2d[16384 + idx] = a1;
  x2d[2 * 16384 + idx] = a2;
}

// 3x3 conv (3ch -> 1ch, SAME) + bias, then concat with y -> xy[2][128][128]
__global__ __launch_bounds__(128) void k_conv2d_xy(
    const float* __restrict__ x2d, const float* __restrict__ w2,
    const float* __restrict__ b2, const float* __restrict__ yimg,
    float* __restrict__ xy)
{
  int idx = blockIdx.x * blockDim.x + threadIdx.x;
  if (idx >= 16384) return;
  int hh = idx >> 7, ww = idx & 127;
  float a = 0.f;
#pragma unroll
  for (int c = 0; c < 3; ++c)
#pragma unroll
    for (int dy = -1; dy <= 1; ++dy)
#pragma unroll
      for (int dx = -1; dx <= 1; ++dx) {
        int yy = hh + dy, xx = ww + dx;
        if (((unsigned)yy < 128u) && ((unsigned)xx < 128u))
          a += x2d[c * 16384 + (yy << 7) + xx] * w2[c * 9 + (dy + 1) * 3 + (dx + 1)];
      }
  xy[idx] = a + b2[0];
  xy[16384 + idx] = yimg[idx];
}

// Patch extraction: out[i][j], feature order (py, px, c) as in the reference.
__global__ void k_patchify(const float* __restrict__ xy, float* __restrict__ out,
                           int patch, int pd, int wq, int total)
{
  int idx = blockIdx.x * blockDim.x + threadIdx.x;
  if (idx >= total) return;
  int i = idx / pd, j = idx % pd;
  int py = j / (patch * 2);
  int rem = j % (patch * 2);
  int px = rem >> 1, c = rem & 1;
  int gy = (i / wq) * patch + py;
  int gx = (i % wq) * patch + px;
  out[idx] = xy[c * 16384 + (gy << 7) + gx];
}

// Row LayerNorm, one block per row.
__global__ __launch_bounds__(256) void k_ln(
    const float* __restrict__ in, float* __restrict__ out,
    const float* __restrict__ g, const float* __restrict__ b, int d)
{
  __shared__ float sh[256];
  const int t = threadIdx.x;
  const float* x = in + (size_t)blockIdx.x * d;
  float* o = out + (size_t)blockIdx.x * d;

  float s = 0.f;
  for (int i = t; i < d; i += 256) s += x[i];
  sh[t] = s; __syncthreads();
  for (int k = 128; k > 0; k >>= 1) { if (t < k) sh[t] += sh[t + k]; __syncthreads(); }
  float mean = sh[0] / d; __syncthreads();

  float v = 0.f;
  for (int i = t; i < d; i += 256) { float e = x[i] - mean; v += e * e; }
  sh[t] = v; __syncthreads();
  for (int k = 128; k > 0; k >>= 1) { if (t < k) sh[t] += sh[t + k]; __syncthreads(); }
  float inv = rsqrtf(sh[0] / d + 1e-5f);

  for (int i = t; i < d; i += 256)
    o[i] = (x[i] - mean) * inv * g[i] + b[i];
}

// Row softmax (in place), one block per row; x *= scale before softmax.
__global__ __launch_bounds__(128) void k_softmax(float* __restrict__ x, int n, float scale)
{
  __shared__ float sh[128];
  const int t = threadIdx.x;
  float* row = x + (size_t)blockIdx.x * n;

  float m = -1e30f;
  for (int i = t; i < n; i += 128) m = fmaxf(m, row[i] * scale);
  sh[t] = m; __syncthreads();
  for (int k = 64; k > 0; k >>= 1) { if (t < k) sh[t] = fmaxf(sh[t], sh[t + k]); __syncthreads(); }
  m = sh[0]; __syncthreads();

  float s = 0.f;
  for (int i = t; i < n; i += 128) { float e = expf(row[i] * scale - m); row[i] = e; s += e; }
  sh[t] = s; __syncthreads();
  for (int k = 64; k > 0; k >>= 1) { if (t < k) sh[t] += sh[t + k]; __syncthreads(); }
  float inv = 1.f / sh[0];
  for (int i = t; i < n; i += 128) row[i] *= inv;
}

__global__ void k_copy(float* __restrict__ d, const float* __restrict__ s, int n) {
  int i = blockIdx.x * blockDim.x + threadIdx.x;
  if (i < n) d[i] = s[i];
}
__global__ void k_add(const float* __restrict__ a, const float* __restrict__ b,
                      float* __restrict__ c, int n) {
  int i = blockIdx.x * blockDim.x + threadIdx.x;
  if (i < n) c[i] = a[i] + b[i];
}
// tokens: row0 = cls, rows >=1 already filled; all rows += pos
__global__ void k_clspos(float* __restrict__ tok, const float* __restrict__ cls,
                         const float* __restrict__ pos, int dim, int total) {
  int i = blockIdx.x * blockDim.x + threadIdx.x;
  if (i >= total) return;
  float v = (i < dim) ? cls[i] : tok[i];
  tok[i] = v + pos[i];
}

// ===========================================================================
// Host-side orchestration
// ===========================================================================
struct Ctx {
  hipStream_t stream;
  char* ws;
  size_t off;
  float* alloc(size_t elems) {
    float* p = (float*)(ws + off);
    off += ((elems * sizeof(float)) + 255) & ~(size_t)255;
    return p;
  }
};

static void gemm(Ctx& c, const float* A, int lda, const float* B, int ldb,
                 const float* bias, const float* Res, int ldr,
                 float* C, int ldc, int M, int N, int K, int flags) {
  dim3 g((M + 63) / 64, (N + 15) / 16);
  k_gemm<<<g, 128, 0, c.stream>>>(A, lda, B, ldb, bias, Res, ldr, C, ldc, M, N, K, flags);
}
static void copyf(Ctx& c, float* d, const float* s, int n) {
  k_copy<<<(n + 255) / 256, 256, 0, c.stream>>>(d, s, n);
}

struct Cursor { void* const* in; int i;
  const float* next() { return (const float*)in[i++]; } };
struct LnP  { const float *g, *b; };
struct LinP { const float *w, *b; };
struct AttnP { LnP ln; const float *wq, *wkv, *wo, *bo; };
struct FfP  { LnP ln; LinP l1, l2; };
struct TransP { int depth; AttnP at[3]; FfP ff[3]; LnP ln; };
struct EmbP { LnP ln1; LinP lin; LnP ln2; const float *pos, *cls; };
struct CrossP { LinP sm_in; AttnP sm_attn; LinP sm_out;
                LinP lg_in; AttnP lg_attn; LinP lg_out; };
struct MsP { TransP sm_enc, lg_enc; CrossP cross[2]; };

static LnP  ln_load(Cursor& c)  { LnP p;  p.g = c.next(); p.b = c.next(); return p; }
static LinP lin_load(Cursor& c) { LinP p; p.w = c.next(); p.b = c.next(); return p; }
static AttnP attn_load(Cursor& c) {
  AttnP p; p.ln = ln_load(c);
  p.wq = c.next(); p.wkv = c.next(); p.wo = c.next(); p.bo = c.next(); return p;
}
static FfP ff_load(Cursor& c) { FfP p; p.ln = ln_load(c); p.l1 = lin_load(c); p.l2 = lin_load(c); return p; }
static TransP trans_load(Cursor& c, int depth) {
  TransP t; t.depth = depth;
  for (int d = 0; d < depth; ++d) { t.at[d] = attn_load(c); t.ff[d] = ff_load(c); }
  t.ln = ln_load(c); return t;
}
static EmbP emb_load(Cursor& c) {
  EmbP e; e.ln1 = ln_load(c); e.lin = lin_load(c); e.ln2 = ln_load(c);
  e.pos = c.next(); e.cls = c.next(); return e;
}
static CrossP cross_load(Cursor& c) {
  CrossP p;
  p.sm_in = lin_load(c); p.sm_attn = attn_load(c); p.sm_out = lin_load(c);
  p.lg_in = lin_load(c); p.lg_attn = attn_load(c); p.lg_out = lin_load(c);
  return p;
}

// Multi-head attention; inner=256, heads=4, dh=64. Returns o@wo+bo (+res).
static void attn_run(Ctx& c, const float* x, int N, int dim, const AttnP& p,
                     const float* context, int Nctx, bool kv_self,
                     const float* res, float* out) {
  const int inner = 256, heads = 4, dh = 64;
  float* xn = c.alloc((size_t)N * dim);
  k_ln<<<N, 256, 0, c.stream>>>(x, xn, p.ln.g, p.ln.b, dim);

  const float* ctx; int Nc;
  if (!context) { ctx = xn; Nc = N; }
  else if (kv_self) {
    Nc = N + Nctx;
    float* cb = c.alloc((size_t)Nc * dim);
    copyf(c, cb, xn, N * dim);
    copyf(c, cb + (size_t)N * dim, context, Nctx * dim);
    ctx = cb;
  } else { ctx = context; Nc = Nctx; }

  float* q = c.alloc((size_t)N * inner);
  gemm(c, xn, dim, p.wq, inner, nullptr, nullptr, 0, q, inner, N, inner, dim, 0);
  float* kv = c.alloc((size_t)Nc * 2 * inner);
  gemm(c, ctx, dim, p.wkv, 2 * inner, nullptr, nullptr, 0, kv, 2 * inner, Nc, 2 * inner, dim, 0);

  float* sc = c.alloc((size_t)N * Nc);
  float* o  = c.alloc((size_t)N * inner);
  for (int h = 0; h < heads; ++h) {
    // scores = q_h @ k_h^T   (K rows are contiguous -> TRANSB path)
    gemm(c, q + h * dh, inner, kv + h * dh, 2 * inner, nullptr, nullptr, 0,
         sc, Nc, N, Nc, dh, GEMM_TRANSB);
    k_softmax<<<N, 128, 0, c.stream>>>(sc, Nc, 0.125f);  // dh^-0.5
    // o_h = scores @ v_h
    gemm(c, sc, Nc, kv + inner + h * dh, 2 * inner, nullptr, nullptr, 0,
         o + h * dh, inner, N, dh, Nc, 0);
  }
  int fl = GEMM_BIAS | (res ? GEMM_RES : 0);
  gemm(c, o, inner, p.wo, dim, p.bo, res, dim, out, dim, N, dim, inner, fl);
}

static float* transformer_run(Ctx& c, float* x, int N, int dim, const TransP& tp) {
  for (int d = 0; d < tp.depth; ++d) {
    float* x1 = c.alloc((size_t)N * dim);
    attn_run(c, x, N, dim, tp.at[d], nullptr, 0, false, x, x1);
    float* xn = c.alloc((size_t)N * dim);
    k_ln<<<N, 256, 0, c.stream>>>(x1, xn, tp.ff[d].ln.g, tp.ff[d].ln.b, dim);
    float* hm = c.alloc((size_t)N * 256);
    gemm(c, xn, dim, tp.ff[d].l1.w, 256, tp.ff[d].l1.b, nullptr, 0,
         hm, 256, N, 256, dim, GEMM_BIAS | GEMM_GELU);
    float* x2 = c.alloc((size_t)N * dim);
    gemm(c, hm, 256, tp.ff[d].l2.w, dim, tp.ff[d].l2.b, x1, dim,
         x2, dim, N, dim, 256, GEMM_BIAS | GEMM_RES);
    x = x2;
  }
  float* xo = c.alloc((size_t)N * dim);
  k_ln<<<N, 256, 0, c.stream>>>(x, xo, tp.ln.g, tp.ln.b, dim);
  return xo;
}

static float* embed_run(Ctx& c, const float* xy, const EmbP& e, int patch, int dim) {
  int wq = 128 / patch, np = wq * wq, pd = patch * patch * 2;
  int tot = np * pd;
  float* pat = c.alloc((size_t)tot);
  k_patchify<<<(tot + 255) / 256, 256, 0, c.stream>>>(xy, pat, patch, pd, wq, tot);
  float* patn = c.alloc((size_t)tot);
  k_ln<<<np, 256, 0, c.stream>>>(pat, patn, e.ln1.g, e.ln1.b, pd);
  float* emb = c.alloc((size_t)np * dim);
  gemm(c, patn, pd, e.lin.w, dim, e.lin.b, nullptr, 0, emb, dim, np, dim, pd, GEMM_BIAS);
  float* tok = c.alloc((size_t)(np + 1) * dim);
  k_ln<<<np, 256, 0, c.stream>>>(emb, tok + dim, e.ln2.g, e.ln2.b, dim);
  int ttot = (np + 1) * dim;
  k_clspos<<<(ttot + 127) / 128, 128, 0, c.stream>>>(tok, e.cls, e.pos, dim, ttot);
  return tok;
}

extern "C" void kernel_launch(void* const* d_in, const int* in_sizes, int n_in,
                              void* d_out, int out_size, void* d_ws, size_t ws_size,
                              hipStream_t stream) {
  (void)in_sizes; (void)n_in; (void)out_size; (void)ws_size;
  Ctx c{stream, (char*)d_ws, 0};

  const float* x      = (const float*)d_in[0];  // (1,1,128,128,128)
  const float* yimg   = (const float*)d_in[1];  // (1,1,128,128)
  const float* theta  = (const float*)d_in[2];  // (1,6)
  const float* corner = (const float*)d_in[3];  // (1,8,4)
  const float* prm    = (const float*)d_in[4];  // (4,)

  // ---- weights, dict-insertion leaf order ----
  Cursor cur{d_in, 5};
  struct { const float *w, *b, *bg, *bb, *bm, *bv; } c3[5];
  for (int i = 0; i < 5; ++i) {
    c3[i].w = cur.next(); c3[i].b = cur.next();
    if (i < 4) { c3[i].bg = cur.next(); c3[i].bb = cur.next();
                 c3[i].bm = cur.next(); c3[i].bv = cur.next(); }
    else { c3[i].bg = c3[i].bb = c3[i].bm = c3[i].bv = nullptr; }
  }
  const float* w2 = cur.next();
  const float* b2 = cur.next();
  EmbP sm_emb = emb_load(cur);
  EmbP lg_emb = emb_load(cur);
  MsP ms[3];
  for (int L = 0; L < 3; ++L) {
    ms[L].sm_enc = trans_load(cur, 2);
    ms[L].lg_enc = trans_load(cur, 3);
    ms[L].cross[0] = cross_load(cur);
    ms[L].cross[1] = cross_load(cur);
  }
  LnP  smh_ln = ln_load(cur);  LinP smh_lin = lin_load(cur);
  LnP  lgh_ln = ln_load(cur);  LinP lgh_lin = lin_load(cur);
  LinP outp = lin_load(cur);

  // ---- conv3d stack (WMMA implicit GEMM + double-buffered TDM weights) ----
  float* bufA = c.alloc((size_t)16 << 21);
  float* bufB = c.alloc((size_t)16 << 21);
  const int cinA[5] = {1, 4, 8, 16, 8};
  const int coutA[5] = {4, 8, 16, 8, 3};
  const float* srcs[5] = {x, bufA, bufB, bufA, bufB};
  float* dsts[5] = {bufA, bufB, bufA, bufB, bufA};
  for (int i = 0; i < 5; ++i)
    k_conv3d<<<131072, 32, 0, stream>>>(
        srcs[i], cinA[i], c3[i].w, c3[i].b, coutA[i],
        c3[i].bg, c3[i].bb, c3[i].bm, c3[i].bv,
        (i == 4) ? x : nullptr, dsts[i]);
  float* x3d = bufA;  // 3 x 128^3, x3d = repeat(x,3)+h fused into last conv

  // ---- projector + 2D conv + channel concat ----
  float* x2d3 = c.alloc(3 * 16384);
  k_project<<<128, 128, 0, stream>>>(x3d, theta, corner, prm, x2d3);
  float* xy = c.alloc(2 * 16384);
  k_conv2d_xy<<<128, 128, 0, stream>>>(x2d3, w2, b2, yimg, xy);

  // ---- CrossViT ----
  float* sm = embed_run(c, xy, sm_emb, 8, 16);    // 257 x 16
  float* lg = embed_run(c, xy, lg_emb, 64, 64);   // 5 x 64
  for (int L = 0; L < 3; ++L) {
    sm = transformer_run(c, sm, 257, 16, ms[L].sm_enc);
    lg = transformer_run(c, lg, 5, 64, ms[L].lg_enc);
    float* smcls = c.alloc(16); copyf(c, smcls, sm, 16);
    float* lgcls = c.alloc(64); copyf(c, lgcls, lg, 64);
    float* sm_p = sm + 16;   // 256 x 16
    float* lg_p = lg + 64;   // 4 x 64
    for (int t = 0; t < 2; ++t) {
      const CrossP& cl = ms[L].cross[t];
      // small cls -> large space, cross-attend over lg patches (+self)
      float* t64 = c.alloc(64);
      gemm(c, smcls, 16, cl.sm_in.w, 64, cl.sm_in.b, nullptr, 0, t64, 64, 1, 64, 16, GEMM_BIAS);
      float* a64 = c.alloc(64);
      attn_run(c, t64, 1, 64, cl.sm_attn, lg_p, 4, true, nullptr, a64);
      float* ns = c.alloc(16);
      gemm(c, a64, 64, cl.sm_out.w, 16, cl.sm_out.b, smcls, 16, ns, 16, 1, 16, 64,
           GEMM_BIAS | GEMM_RES);
      // large cls -> small space, cross-attend over sm patches (+self)
      float* t16 = c.alloc(16);
      gemm(c, lgcls, 64, cl.lg_in.w, 16, cl.lg_in.b, nullptr, 0, t16, 16, 1, 16, 64, GEMM_BIAS);
      float* a16 = c.alloc(16);
      attn_run(c, t16, 1, 16, cl.lg_attn, sm_p, 256, true, nullptr, a16);
      float* nl = c.alloc(64);
      gemm(c, a16, 16, cl.lg_out.w, 64, cl.lg_out.b, lgcls, 64, nl, 64, 1, 64, 16,
           GEMM_BIAS | GEMM_RES);
      smcls = ns; lgcls = nl;
    }
    copyf(c, sm, smcls, 16);
    copyf(c, lg, lgcls, 64);
  }

  // ---- heads + final linear ----
  float* h16 = c.alloc(16);
  k_ln<<<1, 256, 0, stream>>>(sm, h16, smh_ln.g, smh_ln.b, 16);
  float* ls = c.alloc(1000);
  gemm(c, h16, 16, smh_lin.w, 1000, smh_lin.b, nullptr, 0, ls, 1000, 1, 1000, 16, GEMM_BIAS);
  float* h64 = c.alloc(64);
  k_ln<<<1, 256, 0, stream>>>(lg, h64, lgh_ln.g, lgh_ln.b, 64);
  float* ll = c.alloc(1000);
  gemm(c, h64, 64, lgh_lin.w, 1000, lgh_lin.b, nullptr, 0, ll, 1000, 1, 1000, 64, GEMM_BIAS);
  float* logits = c.alloc(1000);
  k_add<<<(1000 + 255) / 256, 256, 0, stream>>>(ls, ll, logits, 1000);
  gemm(c, logits, 1000, outp.w, 1, outp.b, nullptr, 0,
       (float*)d_out, 1, 1, 1, 1000, GEMM_BIAS);
}
